// FlashAttention3_86818468921806
// MI455X (gfx1250) — matrix operand order repaired and checked by reading, not compiled, not run on hardware
//
#include <hip/hip_runtime.h>
#include <hip/hip_bf16.h>

typedef __attribute__((ext_vector_type(16))) __bf16 v16bf;
typedef __attribute__((ext_vector_type(8)))  float  v8f;

#define B_  4
#define S_  2048
#define D_  1024
#define H_  16
#define DH_ 64
#define RSCALE 0.125f   // 1/sqrt(DH)

__device__ __forceinline__ v8f wmma_bf16(v16bf a, v16bf b, v8f c) {
  // D = A(16x32 bf16) x B(32x16 bf16) + C(16x16 f32)
  return __builtin_amdgcn_wmma_f32_16x16x32_bf16(false, a, false, b,
                                                 (short)0, c, false, false);
}

// A-operand (16x32 bf16): lane m holds 16 contiguous K values:
// K = g*16..g*16+15   (g = lane/16)
__device__ __forceinline__ v16bf load_a_frag(const __bf16* row, int g) {
  union { uint4 u[2]; v16bf v; } f;
  f.u[0] = *(const uint4*)(row + g * 16);
  f.u[1] = *(const uint4*)(row + g * 16 + 8);
  return f.v;
}

// B-operand (32x16 bf16): lane holds 16 contiguous K values (32 bytes)
__device__ __forceinline__ v16bf load_b_frag(const __bf16* p) {
  return *(const v16bf*)p;
}

// ---------------- prep: f32 -> bf16 conversions ----------------

__global__ void cvt_x_bf16(const float* __restrict__ in, __bf16* __restrict__ out, int n) {
  int stride = gridDim.x * blockDim.x;
  for (int i = blockIdx.x * blockDim.x + threadIdx.x; i < n; i += stride)
    out[i] = (__bf16)in[i];
}

// out[n*D + k] = (bf16) in[k*D + n]   (W is (Din,Dout); store Wt = (Dout,Din))
__global__ void transpose_w_bf16(const float* __restrict__ in, __bf16* __restrict__ out) {
  int stride = gridDim.x * blockDim.x;
  const int n = D_ * D_;
  for (int i = blockIdx.x * blockDim.x + threadIdx.x; i < n; i += stride) {
    int r = i >> 10;        // output row (= column of W)
    int c = i & (D_ - 1);   // output col (= K)
    out[i] = (__bf16)in[c * D_ + r];
  }
}

// ---------------- fused QKV projection: 64x64 tile per wave ----------------
// grid: (D/64, B*S/64, 3), block: 32

__global__ __launch_bounds__(32)
void proj_qkv(const __bf16* __restrict__ xbf,
              const __bf16* __restrict__ wtq, const __bf16* __restrict__ wtk,
              const __bf16* __restrict__ wtv,
              const float* __restrict__ bq, const float* __restrict__ bk,
              const float* __restrict__ bv,
              __bf16* __restrict__ Q, __bf16* __restrict__ K, __bf16* __restrict__ Vt) {
  const int l = threadIdx.x;
  const int n = l & 15, g = l >> 4;
  const int nt = blockIdx.x, mt = blockIdx.y, which = blockIdx.z;
  const __bf16* wt   = (which == 0) ? wtq : (which == 1) ? wtk : wtv;
  const float*  bias = (which == 0) ? bq  : (which == 1) ? bk  : bv;

  const int row0 = mt * 64;
  const int col0 = nt * 64;

  const __bf16* arow[4];
  const __bf16* brow[4];
#pragma unroll
  for (int i = 0; i < 4; ++i) {
    arow[i] = xbf + (size_t)(row0 + i * 16 + n) * D_;
    brow[i] = wt  + (size_t)(col0 + i * 16 + n) * D_;
  }

  v8f acc[4][4] = {};
  for (int k0 = 0; k0 < D_; k0 += 32) {
    if (k0 + 128 < D_) {
#pragma unroll
      for (int i = 0; i < 4; ++i) {
        __builtin_prefetch(arow[i] + k0 + 128, 0, 3);
        __builtin_prefetch(brow[i] + k0 + 128, 0, 3);
      }
    }
    v16bf af[4], bf[4];
#pragma unroll
    for (int i = 0; i < 4; ++i) af[i] = load_a_frag(arow[i] + k0, g);
#pragma unroll
    for (int i = 0; i < 4; ++i) bf[i] = load_b_frag(brow[i] + k0 + g * 16);
#pragma unroll
    for (int mi = 0; mi < 4; ++mi)
#pragma unroll
      for (int ni = 0; ni < 4; ++ni)
        acc[mi][ni] = wmma_bf16(af[mi], bf[ni], acc[mi][ni]);
  }

  const int bidx = row0 / S_;        // 64-row tile never crosses batch boundary
#pragma unroll
  for (int ni = 0; ni < 4; ++ni) {
    const int col = col0 + ni * 16 + n;
    const float bia = bias[col];
    const int hh = col >> 6, dh = col & 63;
#pragma unroll
    for (int mi = 0; mi < 4; ++mi)
#pragma unroll
      for (int v = 0; v < 8; ++v) {
        const int s = row0 + mi * 16 + v + 8 * g - bidx * S_;
        const float y = acc[mi][ni][v] + bia;
        if (which == 2) {            // V stored transposed: (b,h,dh,s)
          Vt[(((size_t)bidx * H_ + hh) * DH_ + dh) * S_ + s] = (__bf16)y;
        } else if (which == 1) {
          K[(((size_t)bidx * H_ + hh) * S_ + s) * DH_ + dh] = (__bf16)y;
        } else {
          Q[(((size_t)bidx * H_ + hh) * S_ + s) * DH_ + dh] = (__bf16)y;
        }
      }
  }
}

// ---------------- flash attention ----------------
// 4 waves / block, 32 queries per wave (128 queries per block).
// K and V^T 64-key blocks are staged into double-buffered LDS via
// GLOBAL_LOAD_ASYNC_TO_LDS_B128 (ASYNCcnt) and shared by all 4 waves.
// grid: (S/128, H, B), block: 128.

__device__ __forceinline__ void async_copy_kv(const __bf16* __restrict__ Kb,
                                              const __bf16* __restrict__ Vb,
                                              int kb0,
                                              __bf16 (*kdst)[DH_],
                                              __bf16 (*vdst)[64],
                                              int tid) {
#pragma unroll
  for (int i = 0; i < 4; ++i) {
    const int c   = i * 128 + tid;      // 16-byte chunk id (512 per 8KB block)
    const int row = c >> 3;             // 8 chunks per 64-elem row
    const int col = (c & 7) * 8;
    // K block: rows = key tokens, 64 x 64 bf16
    {
      unsigned lds = (unsigned)(size_t)&kdst[row][col];
      const __bf16* gp = Kb + (size_t)(kb0 + row) * DH_ + col;
      asm volatile("global_load_async_to_lds_b128 %0, %1, off"
                   :: "v"(lds), "v"(gp) : "memory");
    }
    // V^T block: rows = dh, cols = key tokens kb0..kb0+63
    {
      unsigned lds = (unsigned)(size_t)&vdst[row][col];
      const __bf16* gp = Vb + (size_t)row * S_ + kb0 + col;
      asm volatile("global_load_async_to_lds_b128 %0, %1, off"
                   :: "v"(lds), "v"(gp) : "memory");
    }
  }
}

__global__ __launch_bounds__(128)
void flash_attn(const __bf16* __restrict__ Q, const __bf16* __restrict__ K,
                const __bf16* __restrict__ Vt, __bf16* __restrict__ O) {
  __shared__ __bf16 kbuf[2][64][DH_];   // 2 x 8KB
  __shared__ __bf16 vbuf[2][DH_][64];   // 2 x 8KB
  __shared__ float  st[4][32][64];      // per-wave score scratch
  __shared__ __bf16 pbf[4][32][64];     // per-wave P tile
  __shared__ float  rowC[4][32];        // per-wave O-rescale coeff

  const int tid = threadIdx.x;
  const int w   = tid >> 5;             // wave id 0..3
  const int l   = tid & 31;             // lane
  const int n   = l & 15, g = l >> 4;
  const int h = blockIdx.y, b = blockIdx.z;
  const size_t bh = (size_t)b * H_ + h;
  const __bf16* Qb = Q  + bh * S_ * DH_;
  const __bf16* Kb = K  + bh * S_ * DH_;
  const __bf16* Vb = Vt + bh * DH_ * S_;

  const int q0wg = blockIdx.x * 128;    // block's first query row
  const int q0   = q0wg + w * 32;       // this wave's first query row

  // Q A-fragments: 32x64 -> 2 query sub-tiles x 2 dh chunks (reused all blocks)
  v16bf qa[2][2];
#pragma unroll
  for (int t = 0; t < 2; ++t) {
    const __bf16* qrow = Qb + (size_t)(q0 + t * 16 + n) * DH_;
    qa[t][0] = load_a_frag(qrow, g);
    qa[t][1] = load_a_frag(qrow + 32, g);
  }

  v8f o[2][4] = {};                     // O accum: 32 queries x 64 dh
  float m_prev = -__builtin_inff();     // lane l owns softmax row l
  float l_prev = 0.f;
  const int r  = l;
  const int qi = q0 + r;

  const int jmax_w  = (q0 + 31) / 64;       // this wave's causal limit
  const int jmaxwg  = (q0wg + 127) / 64;    // block-wide limit (uniform)

  // prologue: prefill both LDS buffers
  async_copy_kv(Kb, Vb, 0, kbuf[0], vbuf[0], tid);
  if (jmaxwg >= 1) async_copy_kv(Kb, Vb, 64, kbuf[1], vbuf[1], tid);

  for (int j = 0; j <= jmaxwg; ++j) {
    const int cur = j & 1;
    const int kb0 = j * 64;

    // retire the async batch for block j (batches complete in order;
    // at most one newer 8-op batch may remain outstanding)
    if (j < jmaxwg) asm volatile("s_wait_asynccnt 0x8" ::: "memory");
    else            asm volatile("s_wait_asynccnt 0x0" ::: "memory");
    __syncthreads();                    // buf[cur] visible to all waves

    if (j <= jmax_w) {                  // wave-uniform causal skip
      // S = (Q K^T) * RSCALE -> per-wave LDS scratch
#pragma unroll
      for (int kt = 0; kt < 4; ++kt) {
        const __bf16* krow = &kbuf[cur][kt * 16 + n][0];
        v16bf kf0 = load_b_frag(krow + g * 16);        // dh 0..31
        v16bf kf1 = load_b_frag(krow + 32 + g * 16);   // dh 32..63
#pragma unroll
        for (int t = 0; t < 2; ++t) {
          v8f s = {};
          s = wmma_bf16(qa[t][0], kf0, s);
          s = wmma_bf16(qa[t][1], kf1, s);
#pragma unroll
          for (int v = 0; v < 8; ++v)
            st[w][t * 16 + v + 8 * g][kt * 16 + n] = s[v] * RSCALE;
        }
      }
      // Online softmax: lane l owns row l (intra-wave DScnt ordering only)
      float mnew = m_prev;
      for (int c = 0; c < 64; ++c) {
        float vv = st[w][r][c];
        if (kb0 + c > qi) vv = -__builtin_inff();      // causal mask
        mnew = fmaxf(mnew, vv);
      }
      const float alpha = __expf(m_prev - mnew);
      float psum = 0.f;
      for (int c = 0; c < 64; ++c) {
        float vv = st[w][r][c];
        if (kb0 + c > qi) vv = -__builtin_inff();
        const float p = __expf(vv - mnew);
        psum += p;
        pbf[w][r][c] = (__bf16)p;
      }
      const float lnew = alpha * l_prev + psum;
      // reference: O_new = O * (l/l_new) * alpha + P@V   (unnormalized)
      rowC[w][r] = (lnew > 0.f) ? (l_prev / lnew) * alpha : 0.f;
      m_prev = mnew;
      l_prev = lnew;

      // O = O*coef + P @ V
#pragma unroll
      for (int t = 0; t < 2; ++t)
#pragma unroll
        for (int v = 0; v < 8; ++v) {
          const float cf = rowC[w][t * 16 + v + 8 * g];
          o[t][0][v] *= cf; o[t][1][v] *= cf; o[t][2][v] *= cf; o[t][3][v] *= cf;
        }
      v16bf pa[2][2];
#pragma unroll
      for (int t = 0; t < 2; ++t) {
        pa[t][0] = load_a_frag(&pbf[w][t * 16 + n][0], g);    // keys 0..31
        pa[t][1] = load_a_frag(&pbf[w][t * 16 + n][32], g);   // keys 32..63
      }
#pragma unroll
      for (int dt = 0; dt < 4; ++dt) {
        const __bf16* vr = &vbuf[cur][dt * 16 + n][0];
        v16bf vf0 = load_b_frag(vr + g * 16);
        v16bf vf1 = load_b_frag(vr + 32 + g * 16);
#pragma unroll
        for (int t = 0; t < 2; ++t) {
          o[t][dt] = wmma_bf16(pa[t][0], vf0, o[t][dt]);
          o[t][dt] = wmma_bf16(pa[t][1], vf1, o[t][dt]);
        }
      }
    }

    __syncthreads();                    // all waves done reading buf[cur]
    if (j + 2 <= jmaxwg)                // refill the freed buffer
      async_copy_kv(Kb, Vb, (j + 2) * 64, kbuf[cur], vbuf[cur], tid);
  }

  // store attention output as bf16 in (B,S,D) layout for the final GEMM
#pragma unroll
  for (int t = 0; t < 2; ++t)
#pragma unroll
    for (int dt = 0; dt < 4; ++dt)
#pragma unroll
      for (int v = 0; v < 8; ++v) {
        const int m = t * 16 + v + 8 * g;
        O[((size_t)b * S_ + q0 + m) * D_ + h * DH_ + dt * 16 + n] =
            (__bf16)o[t][dt][v];
      }
}

// ---------------- output projection: 64x64 tile per wave ----------------
// grid: (D/64, B*S/64), block 32

__global__ __launch_bounds__(32)
void out_proj(const __bf16* __restrict__ A, const __bf16* __restrict__ wt,
              const float* __restrict__ bias, float* __restrict__ out) {
  const int l = threadIdx.x;
  const int n = l & 15, g = l >> 4;
  const int nt = blockIdx.x, mt = blockIdx.y;
  const int row0 = mt * 64;
  const int col0 = nt * 64;

  const __bf16* arow[4];
  const __bf16* brow[4];
#pragma unroll
  for (int i = 0; i < 4; ++i) {
    arow[i] = A  + (size_t)(row0 + i * 16 + n) * D_;
    brow[i] = wt + (size_t)(col0 + i * 16 + n) * D_;
  }

  v8f acc[4][4] = {};
  for (int k0 = 0; k0 < D_; k0 += 32) {
    if (k0 + 128 < D_) {
#pragma unroll
      for (int i = 0; i < 4; ++i) {
        __builtin_prefetch(arow[i] + k0 + 128, 0, 3);
        __builtin_prefetch(brow[i] + k0 + 128, 0, 3);
      }
    }
    v16bf af[4], bf[4];
#pragma unroll
    for (int i = 0; i < 4; ++i) af[i] = load_a_frag(arow[i] + k0, g);
#pragma unroll
    for (int i = 0; i < 4; ++i) bf[i] = load_b_frag(brow[i] + k0 + g * 16);
#pragma unroll
    for (int mi = 0; mi < 4; ++mi)
#pragma unroll
      for (int ni = 0; ni < 4; ++ni)
        acc[mi][ni] = wmma_bf16(af[mi], bf[ni], acc[mi][ni]);
  }

#pragma unroll
  for (int ni = 0; ni < 4; ++ni) {
    const int col = col0 + ni * 16 + n;
    const float bia = bias[col];
#pragma unroll
    for (int mi = 0; mi < 4; ++mi)
#pragma unroll
      for (int v = 0; v < 8; ++v)
        out[(size_t)(row0 + mi * 16 + v + 8 * g) * D_ + col] =
            acc[mi][ni][v] + bia;
  }
}

// ---------------- host launch ----------------

extern "C" void kernel_launch(void* const* d_in, const int* in_sizes, int n_in,
                              void* d_out, int out_size, void* d_ws, size_t ws_size,
                              hipStream_t stream) {
  (void)in_sizes; (void)n_in; (void)out_size; (void)ws_size;

  const float* x  = (const float*)d_in[0];
  const float* Wq = (const float*)d_in[1];
  const float* bq = (const float*)d_in[2];
  const float* Wk = (const float*)d_in[3];
  const float* bk = (const float*)d_in[4];
  const float* Wv = (const float*)d_in[5];
  const float* bv = (const float*)d_in[6];
  const float* Wo = (const float*)d_in[7];
  const float* bo = (const float*)d_in[8];
  float* out = (float*)d_out;

  const size_t n_x = (size_t)B_ * S_ * D_;   // 8,388,608
  const size_t n_w = (size_t)D_ * D_;        // 1,048,576

  char* ws = (char*)d_ws;
  __bf16* xbf = (__bf16*)ws; ws += n_x * 2;
  __bf16* wtq = (__bf16*)ws; ws += n_w * 2;
  __bf16* wtk = (__bf16*)ws; ws += n_w * 2;
  __bf16* wtv = (__bf16*)ws; ws += n_w * 2;
  __bf16* wto = (__bf16*)ws; ws += n_w * 2;
  __bf16* Qb  = (__bf16*)ws; ws += n_x * 2;
  __bf16* Kb  = (__bf16*)ws; ws += n_x * 2;
  __bf16* Vtb = (__bf16*)ws; ws += n_x * 2;
  __bf16* Ob  = (__bf16*)ws; ws += n_x * 2;   // total ~88 MB

  cvt_x_bf16<<<2048, 256, 0, stream>>>(x, xbf, (int)n_x);
  transpose_w_bf16<<<1024, 256, 0, stream>>>(Wq, wtq);
  transpose_w_bf16<<<1024, 256, 0, stream>>>(Wk, wtk);
  transpose_w_bf16<<<1024, 256, 0, stream>>>(Wv, wtv);
  transpose_w_bf16<<<1024, 256, 0, stream>>>(Wo, wto);

  proj_qkv<<<dim3(D_ / 64, (B_ * S_) / 64, 3), 32, 0, stream>>>(
      xbf, wtq, wtk, wtv, bq, bk, bv, Qb, Kb, Vtb);

  flash_attn<<<dim3(S_ / 128, H_, B_), 128, 0, stream>>>(Qb, Kb, Vtb, Ob);

  out_proj<<<dim3(D_ / 64, (B_ * S_) / 64), 32, 0, stream>>>(Ob, wto, bo, out);
}
